// CrossAttention_76227079569555
// MI455X (gfx1250) — matrix operand order repaired and checked by reading, not compiled, not run on hardware
//
#include <hip/hip_runtime.h>

typedef _Float16 half_t;
typedef __attribute__((ext_vector_type(16))) _Float16 v16h;
typedef __attribute__((ext_vector_type(8)))  _Float16 v8h;
typedef __attribute__((ext_vector_type(8)))  float    v8f;

#define NB_B   2
#define NB_N   2048
#define NB_M   2048
#define NB_D   1024
#define NB_H   16
#define NB_DH  64
#define ATT_SCALE 0.125f

#if __has_builtin(__builtin_amdgcn_global_load_async_to_lds_b128) && \
    __has_builtin(__builtin_amdgcn_s_wait_asynccnt)
#define USE_ASYNC 1
#else
#define USE_ASYNC 0
#endif

// ---------------- fragment helpers ----------------
__device__ __forceinline__ v16h frag2(const half_t* lo, const half_t* hi) {
  v8h a = *(const v8h*)lo;
  v8h b = *(const v8h*)hi;
  v16h r;
#pragma unroll
  for (int i = 0; i < 8; ++i) { r[i] = a[i]; r[8 + i] = b[i]; }
  return r;
}

__device__ __forceinline__ v8f wmma_f16(v16h a, v16h b, v8f c) {
  return __builtin_amdgcn_wmma_f32_16x16x32_f16(false, a, false, b, (short)0, c,
                                                false, false);
}

#if USE_ASYNC
// exact pointer types per hipcc diagnostic: v4i in AS1 (global) / AS3 (LDS)
typedef int v4i_ __attribute__((vector_size(16)));
typedef __attribute__((address_space(1))) v4i_ gv4i;
typedef __attribute__((address_space(3))) v4i_ lv4i;

// async 16-byte global -> LDS copy (per lane), tracked by ASYNCcnt
__device__ __forceinline__ void async_ld16(const half_t* g, half_t* l) {
  __builtin_amdgcn_global_load_async_to_lds_b128(
      (gv4i*)(uintptr_t)g, (lv4i*)(uintptr_t)l, 0, 0);
}
#endif

// ---------------- f32 -> f16 convert ----------------
struct __align__(8) H4 { half_t h[4]; };

__global__ void cvt_f32_to_f16(const float* __restrict__ src,
                               half_t* __restrict__ dst, int n4) {
  int i = blockIdx.x * blockDim.x + threadIdx.x;
  if (i < n4) {
    float4 v = ((const float4*)src)[i];
    H4 o;
    o.h[0] = (half_t)v.x; o.h[1] = (half_t)v.y;
    o.h[2] = (half_t)v.z; o.h[3] = (half_t)v.w;
    ((H4*)dst)[i] = o;
  }
}

// ---------------- 1024x1024 transpose + convert (for weights) ----------------
__global__ void transpose_cvt(const float* __restrict__ src,
                              half_t* __restrict__ dst) {
  __shared__ float tile[32][33];
  int bx = blockIdx.x, by = blockIdx.y;
  int tx = threadIdx.x, ty = threadIdx.y;  // 32 x 8
#pragma unroll
  for (int k = 0; k < 32; k += 8)
    tile[ty + k][tx] = src[(by * 32 + ty + k) * NB_D + bx * 32 + tx];
  __syncthreads();
#pragma unroll
  for (int k = 0; k < 32; k += 8)
    dst[(bx * 32 + ty + k) * NB_D + by * 32 + tx] = (half_t)tile[tx][ty + k];
}

// ---------------- WMMA GEMM:  C = alpha * A[Mr,K] * Bt[Nc,K]^T (+bias) ----------
// LDS-staged, double-buffered with async-to-LDS DMA when available.
// MODE 0: store f16 per-head row-major  qh/kh[((b*H+h)*NM + n)*64 + d]
// MODE 1: store f16 per-head TRANSPOSED vT[((b*H+h)*64 + d)*NM + n]
// MODE 2: store f32 row-major + bias    C32[row*Ncols + col]
template <int MODE>
__global__ __launch_bounds__(256) void gemm_wmma(
    const half_t* __restrict__ A, const half_t* __restrict__ Bt,
    half_t* __restrict__ C16, float* __restrict__ C32,
    const float* __restrict__ bias, int K, int Ncols, int NM, float alpha) {
  __shared__ __align__(16) half_t sA[2][64 * 32];   // 64 rows x 32 k
  __shared__ __align__(16) half_t sB[2][128 * 32];  // 128 cols x 32 k

  const int lane = threadIdx.x & 31;
  const int w    = threadIdx.x >> 5;
  const int lm   = lane & 15;
  const int g    = lane >> 4;
  const int wm   = w & 3;       // 4 waves along M
  const int wn   = w >> 2;      // 2 waves along N
  const int row0 = blockIdx.x * 64 + wm * 16;
  const int col0 = blockIdx.y * 128 + wn * 64;

  const int tid = threadIdx.x;
  const int ar = tid >> 2, ac = (tid & 3) * 8;   // A coop: row 0..63, 8 halves
  const int br = tid >> 1, bc = (tid & 1) * 16;  // B coop: col 0..127, 16 halves
  const half_t* Ab = A + (blockIdx.x * 64) * K;
  const half_t* Bb = Bt + (blockIdx.y * 128) * K;

  const int NSTEP = K / 32;
  v8f acc[4] = {};

#if USE_ASYNC
  // prologue: DMA k-tile 0 into buffer 0
  async_ld16(Ab + ar * K + ac, &sA[0][ar * 32 + ac]);
  async_ld16(Bb + br * K + bc, &sB[0][br * 32 + bc]);
  async_ld16(Bb + br * K + bc + 8, &sB[0][br * 32 + bc + 8]);
#endif

  for (int i = 0; i < NSTEP; ++i) {
    const int k0  = i * 32;
    const int cur = i & 1;
#if USE_ASYNC
    __builtin_amdgcn_s_wait_asynccnt(0);
    __syncthreads();
    if (i + 1 < NSTEP) {
      const int kn = k0 + 32;
      async_ld16(Ab + ar * K + kn + ac, &sA[1 - cur][ar * 32 + ac]);
      async_ld16(Bb + br * K + kn + bc, &sB[1 - cur][br * 32 + bc]);
      async_ld16(Bb + br * K + kn + bc + 8, &sB[1 - cur][br * 32 + bc + 8]);
    }
    const half_t* cA = sA[cur];
    const half_t* cB = sB[cur];
#else
    *(v8h*)&sA[0][ar * 32 + ac]     = *(const v8h*)(Ab + ar * K + k0 + ac);
    *(v8h*)&sB[0][br * 32 + bc]     = *(const v8h*)(Bb + br * K + k0 + bc);
    *(v8h*)&sB[0][br * 32 + bc + 8] = *(const v8h*)(Bb + br * K + k0 + bc + 8);
    __syncthreads();
    const half_t* cA = sA[0];
    const half_t* cB = sB[0];
#endif
    const half_t* pa = &cA[(wm * 16 + lm) * 32];
    v16h af = frag2(pa + g * 16, pa + g * 16 + 8);
#pragma unroll
    for (int t = 0; t < 4; ++t) {
      const half_t* pb = &cB[(wn * 64 + t * 16 + lm) * 32 + g * 16];
      v16h bf = frag2(pb, pb + 8);
      acc[t] = wmma_f16(af, bf, acc[t]);
    }
#if !USE_ASYNC
    __syncthreads();
#endif
  }

  const int bidx  = row0 / NM;   // batch index
  const int nbase = row0 % NM;
#pragma unroll
  for (int t = 0; t < 4; ++t) {
    const int ct = col0 + t * 16 + lm;   // lane-fixed output column
    const int h  = ct >> 6;
    const int d  = ct & 63;
    if constexpr (MODE == 0) {
#pragma unroll
      for (int r = 0; r < 8; ++r) {
        int n = nbase + 8 * g + r;
        C16[(((bidx * NB_H + h) * NM + n) << 6) + d] =
            (half_t)(acc[t][r] * alpha);
      }
    } else if constexpr (MODE == 1) {
      v8h o;
#pragma unroll
      for (int r = 0; r < 8; ++r) o[r] = (half_t)(acc[t][r] * alpha);
      *(v8h*)(C16 + ((((bidx * NB_H + h) << 6) + d) * NM + nbase + 8 * g)) = o;
    } else {
      float bv = bias ? bias[ct] : 0.0f;
#pragma unroll
      for (int r = 0; r < 8; ++r) {
        int rr = row0 + 8 * g + r;
        C32[rr * Ncols + ct] = acc[t][r] * alpha + bv;
      }
    }
  }
}

// ---------------- flash attention (S^T formulation) ----------------
// qh[((b*H+h)*N + n)*64 + d]  (scale pre-applied)
// kh[((b*H+h)*M + m)*64 + d]
// vT[((b*H+h)*64 + d)*M + m]
// o16[(b*N + n)*1024 + h*64 + d]
__global__ __launch_bounds__(256) void attn_fwd(const half_t* __restrict__ qh,
                                                const half_t* __restrict__ kh,
                                                const half_t* __restrict__ vT,
                                                half_t* __restrict__ o16) {
  __shared__ __align__(16) half_t sK[2][32 * 64];  // 32 keys x 64 dims
  __shared__ __align__(16) half_t sV[2][64 * 32];  // 64 dims x 32 keys (V^T)

  const int lane = threadIdx.x & 31;
  const int w    = threadIdx.x >> 5;
  const int lm   = lane & 15;
  const int g    = lane >> 4;
  const int h    = blockIdx.y;
  const int b    = blockIdx.z;
  const int bh   = b * NB_H + h;
  const int n0   = blockIdx.x * 128 + w * 16;

  // Q^T B-fragments for this wave's 16 query rows (held in regs for all of M)
  const half_t* qp = qh + (((bh * NB_N) + n0 + lm) << 6);
  v16h bq0 = frag2(qp + g * 8, qp + 16 + g * 8);
  v16h bq1 = frag2(qp + 32 + g * 8, qp + 48 + g * 8);

  float mrun = -3.0e38f, lrun = 0.0f;
  v8f accO[4] = {};

  const int tid = threadIdx.x;
  const int kr = tid >> 3, kc = (tid & 7) * 8;  // K-tile coop load
  const int vr = tid >> 2, vc = (tid & 3) * 8;  // V^T-tile coop load
  const half_t* kbase = kh + ((bh * NB_M) << 6);
  const half_t* vbase = vT + (bh << 6) * NB_M;

  const int NSTEP = NB_M / 32;

#if USE_ASYNC
  // prologue: DMA tile 0 into buffer 0
  async_ld16(kbase + (kr << 6) + kc, &sK[0][kr * 64 + kc]);
  async_ld16(vbase + vr * NB_M + vc, &sV[0][vr * 32 + vc]);
#endif

  for (int i = 0; i < NSTEP; ++i) {
    const int m0  = i * 32;
    const int cur = i & 1;

#if USE_ASYNC
    __builtin_amdgcn_s_wait_asynccnt(0);  // own DMA for tile i complete
    __syncthreads();                      // all waves' DMAs + prev readers done
    if (i + 1 < NSTEP) {                  // overlap tile i+1 DMA with compute
      const int mn = m0 + 32;
      async_ld16(kbase + ((mn + kr) << 6) + kc, &sK[1 - cur][kr * 64 + kc]);
      async_ld16(vbase + vr * NB_M + mn + vc, &sV[1 - cur][vr * 32 + vc]);
    }
    const half_t* cK = sK[cur];
    const half_t* cV = sV[cur];
#else
    *(v8h*)&sK[0][kr * 64 + kc] = *(const v8h*)(kbase + ((m0 + kr) << 6) + kc);
    *(v8h*)&sV[0][vr * 32 + vc] = *(const v8h*)(vbase + vr * NB_M + m0 + vc);
    __syncthreads();
    const half_t* cK = sK[0];
    const half_t* cV = sV[0];
#endif

    // S^T tiles: rows = keys, cols = queries
    v8f st0 = {}, st1 = {};
    {
      const half_t* kp0 = &cK[lm << 6];
      const half_t* kp1 = &cK[(16 + lm) << 6];
      v16h a;
      a = frag2(kp0 + g * 8, kp0 + 16 + g * 8);       st0 = wmma_f16(a, bq0, st0);
      a = frag2(kp0 + 32 + g * 8, kp0 + 48 + g * 8);  st0 = wmma_f16(a, bq1, st0);
      a = frag2(kp1 + g * 8, kp1 + 16 + g * 8);       st1 = wmma_f16(a, bq0, st1);
      a = frag2(kp1 + 32 + g * 8, kp1 + 48 + g * 8);  st1 = wmma_f16(a, bq1, st1);
    }

    // online softmax: each lane owns one query row (n = n0 + lm)
    float tmax = st0[0];
#pragma unroll
    for (int r = 0; r < 8; ++r) {
      tmax = fmaxf(tmax, st0[r]);
      tmax = fmaxf(tmax, st1[r]);
    }
    tmax = fmaxf(tmax, __shfl_xor(tmax, 16, 32));
    float mnew = fmaxf(mrun, tmax);
    float corr = __expf(mrun - mnew);
    float p0[8], p1[8], ls = 0.0f;
#pragma unroll
    for (int r = 0; r < 8; ++r) {
      p0[r] = __expf(st0[r] - mnew);
      p1[r] = __expf(st1[r] - mnew);
      ls += p0[r] + p1[r];
    }
    ls += __shfl_xor(ls, 16, 32);
    lrun = lrun * corr + ls;
    mrun = mnew;
#pragma unroll
    for (int dt = 0; dt < 4; ++dt)
#pragma unroll
      for (int r = 0; r < 8; ++r) accO[dt][r] *= corr;

    // pack P^T into a B-fragment (K = 32 keys) via cross-half exchange
    v16h bp;
#pragma unroll
    for (int r = 0; r < 8; ++r) {
      float o0 = __shfl_xor(p0[r], 16, 32);
      float o1 = __shfl_xor(p1[r], 16, 32);
      bp[r]     = (half_t)(g ? o1 : p0[r]);
      bp[8 + r] = (half_t)(g ? p1[r] : o0);
    }

    // O^T += V^T * P^T
#pragma unroll
    for (int dt = 0; dt < 4; ++dt) {
      const half_t* vp = &cV[(dt * 16 + lm) * 32];
      v16h a = frag2(vp + g * 16, vp + g * 16 + 8);
      accO[dt] = wmma_f16(a, bp, accO[dt]);
    }
#if !USE_ASYNC
    __syncthreads();
#endif
  }

  float inv = (lrun > 0.0f) ? 1.0f / lrun : 0.0f;
  half_t* ob = o16 + (((b * NB_N) + n0 + lm) << 10) + (h << 6);
#pragma unroll
  for (int dt = 0; dt < 4; ++dt) {
#pragma unroll
    for (int rp = 0; rp < 4; ++rp) {
      int d = dt * 16 + 8 * g + 2 * rp;
      union { half_t h2[2]; unsigned u; } pk;
      pk.h2[0] = (half_t)(accO[dt][2 * rp] * inv);
      pk.h2[1] = (half_t)(accO[dt][2 * rp + 1] * inv);
      *(unsigned*)(ob + d) = pk.u;
    }
  }
}

// ---------------- host side ----------------
extern "C" void kernel_launch(void* const* d_in, const int* in_sizes, int n_in,
                              void* d_out, int out_size, void* d_ws,
                              size_t ws_size, hipStream_t stream) {
  (void)in_sizes; (void)n_in; (void)out_size; (void)ws_size;
  const float* x    = (const float*)d_in[0];
  const float* ctx  = (const float*)d_in[1];
  const float* Wq   = (const float*)d_in[2];
  const float* Wk   = (const float*)d_in[3];
  const float* Wv   = (const float*)d_in[4];
  const float* Wo   = (const float*)d_in[5];
  const float* bo   = (const float*)d_in[6];
  float* out        = (float*)d_out;

  const int XE = NB_B * NB_N * NB_D;  // 4194304 elements
  const int WE = NB_D * NB_D;         // 1048576 elements per weight

  half_t* base  = (half_t*)d_ws;
  half_t* x16   = base;
  half_t* c16   = x16 + XE;
  half_t* wqT   = c16 + XE;
  half_t* wkT   = wqT + WE;
  half_t* wvT   = wkT + WE;
  half_t* woT   = wvT + WE;
  half_t* qhh   = woT + WE;
  half_t* khh   = qhh + XE;
  half_t* vTt   = khh + XE;
  half_t* o16   = vTt + XE;

  // 1) convert activations to f16
  cvt_f32_to_f16<<<dim3(XE / 4 / 256), dim3(256), 0, stream>>>(x, x16, XE / 4);
  cvt_f32_to_f16<<<dim3(XE / 4 / 256), dim3(256), 0, stream>>>(ctx, c16, XE / 4);

  // 2) transpose + convert weights
  dim3 tg(32, 32), tb(32, 8);
  transpose_cvt<<<tg, tb, 0, stream>>>(Wq, wqT);
  transpose_cvt<<<tg, tb, 0, stream>>>(Wk, wkT);
  transpose_cvt<<<tg, tb, 0, stream>>>(Wv, wvT);
  transpose_cvt<<<tg, tb, 0, stream>>>(Wo, woT);

  // 3) projections (WMMA)
  dim3 gg(NB_B * NB_N / 64, NB_D / 128), gb(256);
  gemm_wmma<0><<<gg, gb, 0, stream>>>(x16, wqT, qhh, nullptr, nullptr,
                                      NB_D, NB_D, NB_N, ATT_SCALE);
  gemm_wmma<0><<<gg, gb, 0, stream>>>(c16, wkT, khh, nullptr, nullptr,
                                      NB_D, NB_D, NB_M, 1.0f);
  gemm_wmma<1><<<gg, gb, 0, stream>>>(c16, wvT, vTt, nullptr, nullptr,
                                      NB_D, NB_D, NB_M, 1.0f);

  // 4) flash attention
  attn_fwd<<<dim3(NB_N / 128, NB_H, NB_B), dim3(256), 0, stream>>>(qhh, khh,
                                                                   vTt, o16);

  // 5) output projection + bias (f32 out)
  gemm_wmma<2><<<gg, gb, 0, stream>>>(o16, woT, nullptr, out, bo,
                                      NB_D, NB_D, NB_N, 1.0f);
}
